// MyModel_55903294325170
// MI455X (gfx1250) — compile-verified, matrix operand order checked by reading, not run on hardware
//
#include <hip/hip_runtime.h>
#include <hip/hip_bf16.h>

// Problem dims (fixed by reference)
#define BDIM 64
#define TDIM 2048
#define HDIM 1024
#define ADIM 256
#define ROWS (BDIM * TDIM)          // 131072 rows of X

typedef __attribute__((ext_vector_type(16))) __bf16 v16bf;
typedef __attribute__((ext_vector_type(8)))  float  v8f;

// Pointer types matching __builtin_amdgcn_global_load_async_to_lds_b128:
//   (v4i addrspace(1)*, v4i addrspace(3)*, imm offset, imm cpol)
typedef __attribute__((__vector_size__(4 * sizeof(int)))) int async_v4i;
typedef __attribute__((address_space(1))) async_v4i* gas_v4i_ptr;
typedef __attribute__((address_space(3))) async_v4i* las_v4i_ptr;

// ---- CDNA5 feature probes (compile-safe) ----------------------------------
#if defined(__has_builtin)
#  if __has_builtin(__builtin_amdgcn_global_load_async_to_lds_b128)
#    define HAVE_ASYNC 1
#  endif
#  if __has_builtin(__builtin_amdgcn_tanhf)
#    define FAST_TANH(x) __builtin_amdgcn_tanhf(x)
#  endif
#endif
#ifndef HAVE_ASYNC
#  define HAVE_ASYNC 0
#endif
#ifndef FAST_TANH
#  define FAST_TANH(x) tanhf(x)
#endif

__device__ __forceinline__ void wait_async0() {
#if defined(__has_builtin) && __has_builtin(__builtin_amdgcn_s_wait_asynccnt)
    __builtin_amdgcn_s_wait_asynccnt(0);
#else
    asm volatile("s_wait_asynccnt 0x0" ::: "memory");
#endif
}

// K-offset pattern inside a 16-bit WMMA fragment (A and B mirror each other):
// lanes 0-15 hold K = {0..7, 16..23}, lanes 16-31 hold K = {8..15, 24..31},
// element i of the v16 maps to K = (i&7) + 16*(i>>3) + 8*half.
__device__ __forceinline__ int frag_koff(int i, int half) {
    return (i & 7) + ((i >> 3) << 4) + (half << 3);
}

// ---------------------------------------------------------------------------
// Kernel 0: repack W [H=1024, A=256] f32 row-major -> bf16 B-fragments,
// kc-major so one K-step's 16 N-tile fragments are a contiguous 16 KB slice:
//   packed[((kc*16 + nt)*32 + lane)*16 + i] = bf16( W[k, n] )
//   n = nt*16 + (lane&15),  k = kc*32 + frag_koff(i, lane>>4)
// ---------------------------------------------------------------------------
__global__ __launch_bounds__(256)
void packw_kernel(const float* __restrict__ w, __bf16* __restrict__ pw) {
    int idx  = blockIdx.x * 256 + threadIdx.x;   // 0 .. 262143
    int i    = idx & 15;
    int lane = (idx >> 4) & 31;
    int nt   = (idx >> 9) & 15;
    int kc   = idx >> 13;
    int half = lane >> 4;
    int n = nt * 16 + (lane & 15);
    int k = kc * 32 + frag_koff(i, half);
    pw[idx] = (__bf16)w[k * ADIM + n];
}

// ---------------------------------------------------------------------------
// Kernel 1: vu[row] = sum_a tanh( X[row,:] . W[:,a] + b[a] ) * u[a]
// 8 waves / block, 16 rows / wave.  kc outer, nt inner with 16 live f32
// accumulators (X streamed from HBM once).  B slices staged to LDS with
// double-buffered GLOBAL_LOAD_ASYNC_TO_LDS_B128 (ASYNCcnt), overlapping the
// L2->LDS copy of slice kc+1 with the 16 WMMAs of slice kc.
// ---------------------------------------------------------------------------
__global__ __launch_bounds__(256)
void vu_kernel(const float* __restrict__ inp,
               const __bf16* __restrict__ pw,
               const float* __restrict__ bo,
               const float* __restrict__ uo,
               float* __restrict__ vu) {
#if HAVE_ASYNC
    __shared__ uint4 bstage[2][1024];   // 2 x 16 KB double buffer
#else
    __shared__ uint4 bstage[1][1024];
#endif

    const int lane = threadIdx.x & 31;
    const int wave = threadIdx.x >> 5;
    const int half = lane >> 4;      // 0: lanes 0-15, 1: lanes 16-31
    const int l15  = lane & 15;
    const long row0 = (long)blockIdx.x * 128 + wave * 16;

    const float* __restrict__ arow = inp + (row0 + l15) * (long)HDIM;
    const uint4* __restrict__ pwu4 = (const uint4*)pw;
    (void)pwu4;

    v8f acc[16];
#pragma unroll
    for (int nt = 0; nt < 16; ++nt) {
        v8f z = {};
        acc[nt] = z;
    }

#if HAVE_ASYNC
    // Each thread async-copies 64 B of the 16 KB slice: 4 x b128.
    const char* gbase = (const char*)pw + (size_t)threadIdx.x * 16;
    char*       lbase = (char*)&bstage[0][0] + (size_t)threadIdx.x * 16;
#pragma unroll
    for (int j = 0; j < 4; ++j)
        __builtin_amdgcn_global_load_async_to_lds_b128(
            (gas_v4i_ptr)(gbase + j * 4096),
            (las_v4i_ptr)(lbase + j * 4096),
            0, 0);
#endif

    for (int kc = 0; kc < 32; ++kc) {
        const uint4* bbuf;
#if HAVE_ASYNC
        wait_async0();        // this wave's copies into buf[kc&1] are complete
        __syncthreads();      // all waves ready; buf[1-(kc&1)] no longer read
        if (kc + 1 < 32) {
            const char* g = gbase + (size_t)(kc + 1) * 16384;
            char*       l = (char*)&bstage[(kc + 1) & 1][0] + (size_t)threadIdx.x * 16;
#pragma unroll
            for (int j = 0; j < 4; ++j)
                __builtin_amdgcn_global_load_async_to_lds_b128(
                    (gas_v4i_ptr)(g + j * 4096),
                    (las_v4i_ptr)(l + j * 4096),
                    0, 0);
        }
        bbuf = bstage[kc & 1];
#else
        __syncthreads();
        {
            const uint4* src = pwu4 + (size_t)kc * 1024 + threadIdx.x;
#pragma unroll
            for (int j = 0; j < 4; ++j)
                bstage[0][threadIdx.x + j * 256] = src[j * 256];
        }
        __syncthreads();
        bbuf = bstage[0];
#endif

        // ---- A fragment: f32 -> bf16 in-register ----
        const int ka = kc * 32 + half * 8;
        const float4* ap = (const float4*)(arow + ka);
        float4 x0 = ap[0];           // k .. k+3
        float4 x1 = ap[1];           // k+4 .. k+7
        float4 x2 = ap[4];           // k+16 .. k+19
        float4 x3 = ap[5];           // k+20 .. k+23
        __builtin_prefetch(arow + ka + 32, 0, 3);   // global_prefetch_b8

        v16bf a;
        a[0]  = (__bf16)x0.x; a[1]  = (__bf16)x0.y;
        a[2]  = (__bf16)x0.z; a[3]  = (__bf16)x0.w;
        a[4]  = (__bf16)x1.x; a[5]  = (__bf16)x1.y;
        a[6]  = (__bf16)x1.z; a[7]  = (__bf16)x1.w;
        a[8]  = (__bf16)x2.x; a[9]  = (__bf16)x2.y;
        a[10] = (__bf16)x2.z; a[11] = (__bf16)x2.w;
        a[12] = (__bf16)x3.x; a[13] = (__bf16)x3.y;
        a[14] = (__bf16)x3.z; a[15] = (__bf16)x3.w;

        // ---- 16 independent WMMAs: same A, 16 N-tile B fragments from LDS ----
        const __bf16* bl = (const __bf16*)bbuf + (size_t)lane * 16;
#pragma unroll
        for (int nt = 0; nt < 16; ++nt) {
            v16bf bm = *(const v16bf*)(bl + (size_t)nt * 512);
            acc[nt] = __builtin_amdgcn_wmma_f32_16x16x32_bf16(
                false, a, false, bm, (short)0, acc[nt], false, false);
        }
    }

    // ---- epilogue: tanh(+b)*u per C element, reduce over A ----
    // C layout => VGPR j, this lane holds (M = j + 8*half, N = l15)
    float red[8];
#pragma unroll
    for (int j = 0; j < 8; ++j) red[j] = 0.0f;

#pragma unroll
    for (int nt = 0; nt < 16; ++nt) {
        const int   n    = nt * 16 + l15;
        const float bo_n = bo[n];
        const float uo_n = uo[n];
#pragma unroll
        for (int j = 0; j < 8; ++j)
            red[j] += FAST_TANH(acc[nt][j] + bo_n) * uo_n;
    }

    // Reduce over the 16 N-lanes of each half-wave; lane l15==0 writes.
#pragma unroll
    for (int j = 0; j < 8; ++j) {
        float r = red[j];
        r += __shfl_xor(r, 8, 16);
        r += __shfl_xor(r, 4, 16);
        r += __shfl_xor(r, 2, 16);
        r += __shfl_xor(r, 1, 16);
        if (l15 == 0)
            vu[row0 + 8 * half + j] = r;
    }
}

// ---------------------------------------------------------------------------
// Kernel 2: softmax over T for each batch row. One block per b.
// ---------------------------------------------------------------------------
__global__ __launch_bounds__(256)
void softmax_kernel(const float* __restrict__ vu, float* __restrict__ al) {
    __shared__ float sm[TDIM];
    __shared__ float redmax[8];
    __shared__ float redsum[8];

    const int b = blockIdx.x;
    const float* row = vu + (size_t)b * TDIM;

    float lmax = -3.402823466e38f;
    for (int t = threadIdx.x; t < TDIM; t += 256) {
        float v = row[t];
        sm[t] = v;
        lmax = fmaxf(lmax, v);
    }
#pragma unroll
    for (int m = 16; m >= 1; m >>= 1) lmax = fmaxf(lmax, __shfl_xor(lmax, m, 32));
    if ((threadIdx.x & 31) == 0) redmax[threadIdx.x >> 5] = lmax;
    __syncthreads();
    float bmax = redmax[0];
#pragma unroll
    for (int i = 1; i < 8; ++i) bmax = fmaxf(bmax, redmax[i]);

    float lsum = 0.0f;
    for (int t = threadIdx.x; t < TDIM; t += 256) {
        float e = __expf(sm[t] - bmax);
        sm[t] = e;
        lsum += e;
    }
#pragma unroll
    for (int m = 16; m >= 1; m >>= 1) lsum += __shfl_xor(lsum, m, 32);
    if ((threadIdx.x & 31) == 0) redsum[threadIdx.x >> 5] = lsum;
    __syncthreads();
    float s = 0.0f;
#pragma unroll
    for (int i = 0; i < 8; ++i) s += redsum[i];
    const float inv = 1.0f / s;

    for (int t = threadIdx.x; t < TDIM; t += 256)
        al[(size_t)b * TDIM + t] = sm[t] * inv;
}

// ---------------------------------------------------------------------------
// Kernel 3: out[b,h] = sum_t al[b,t] * X[b,t,h]; out[b,H] = 1.
// Grid (H/256, B); coalesced streaming of X, alphas staged in LDS.
// ---------------------------------------------------------------------------
__global__ __launch_bounds__(256)
void pool_kernel(const float* __restrict__ inp,
                 const float* __restrict__ al,
                 float* __restrict__ out) {
    __shared__ float a[TDIM];
    const int b = blockIdx.y;
    for (int t = threadIdx.x; t < TDIM; t += 256)
        a[t] = al[(size_t)b * TDIM + t];
    __syncthreads();

    const int h = blockIdx.x * 256 + threadIdx.x;
    const float* __restrict__ base = inp + (size_t)b * TDIM * HDIM + h;

    float acc = 0.0f;
#pragma unroll 8
    for (int t = 0; t < TDIM; ++t)
        acc = fmaf(a[t], base[(size_t)t * HDIM], acc);

    out[(size_t)b * (HDIM + 1) + h] = acc;
    if (h == 0)
        out[(size_t)b * (HDIM + 1) + HDIM] = 1.0f;
}

// ---------------------------------------------------------------------------
extern "C" void kernel_launch(void* const* d_in, const int* in_sizes, int n_in,
                              void* d_out, int out_size, void* d_ws, size_t ws_size,
                              hipStream_t stream) {
    (void)in_sizes; (void)n_in; (void)out_size; (void)ws_size;

    const float* inp = (const float*)d_in[0];   // [B,T,H]
    const float* w   = (const float*)d_in[1];   // [H,A]
    const float* bo  = (const float*)d_in[2];   // [A]
    const float* uo  = (const float*)d_in[3];   // [A]
    float* out       = (float*)d_out;           // [B, H+1]

    char* ws = (char*)d_ws;
    __bf16* pw = (__bf16*)ws;                         // 512 KB packed W
    float*  vu = (float*)(ws + 512 * 1024);           // 512 KB scores
    float*  al = (float*)(ws + 1024 * 1024);          // 512 KB alphas

    packw_kernel<<<1024, 256, 0, stream>>>(w, pw);
    vu_kernel<<<ROWS / 128, 256, 0, stream>>>(inp, pw, bo, uo, vu);
    softmax_kernel<<<BDIM, 256, 0, stream>>>(vu, al);
    pool_kernel<<<dim3(HDIM / 256, BDIM), 256, 0, stream>>>(inp, al, out);
}
